// ViT_49237505081621
// MI455X (gfx1250) — compile-verified, matrix operand order checked by reading, not run on hardware
//
#include <hip/hip_runtime.h>

// ViT patch-embedding as an implicit-GEMM:
//   out[50176, 768] = patchify(images)[50176, 768(K)] x W^T + b
// f16 WMMA (16x16x32) with fp32 accumulate; W pre-converted to f16 in d_ws.

typedef __attribute__((ext_vector_type(16))) _Float16 v16h;
typedef __attribute__((ext_vector_type(8)))  _Float16 v8h;
typedef __attribute__((ext_vector_type(4)))  _Float16 v4h;
typedef __attribute__((ext_vector_type(8)))  float    v8f;
typedef __attribute__((ext_vector_type(4)))  float    v4f;

#define HD       768          // hidden dim == input dim (3*16*16)
#define PATCHES  196          // 14*14 patches per image
#define IMG_CH   50176        // 224*224 (channel stride, floats)
#define IMG_N    150528       // 3*224*224 (image stride, floats)
#define BM       128          // block tile M
#define BN       128          // block tile N
#define WM       32           // wave tile M
#define WN       64           // wave tile N

__global__ void __launch_bounds__(256)
w_to_f16_kernel(const float* __restrict__ W, _Float16* __restrict__ Wh) {
  int i = (blockIdx.x * 256 + threadIdx.x) * 4;   // 768*768 / 4 threads, exact
  v4f v = *reinterpret_cast<const v4f*>(W + i);
  v4h o;
  o[0] = (_Float16)v[0]; o[1] = (_Float16)v[1];
  o[2] = (_Float16)v[2]; o[3] = (_Float16)v[3];
  *reinterpret_cast<v4h*>(Wh + i) = o;
}

__global__ void __launch_bounds__(256)
vit_patch_embed(const float* __restrict__ img, const _Float16* __restrict__ Wh,
                const float* __restrict__ bias, float* __restrict__ out) {
  const int lane  = threadIdx.x & 31;
  const int wave  = threadIdx.x >> 5;    // 8 waves: 4 along M x 2 along N
  const int wm    = wave & 3;
  const int wn    = wave >> 1 >> 1;      // wave >> 2
  const int l15   = lane & 15;
  const int lhalf = lane >> 4;           // K-chunk half per WMMA A/B layout

  const int m_base = blockIdx.y * BM + wm * WM;
  const int n_base = blockIdx.x * BN + wn * WN;

  // Per-lane image base (element index) for the two 16-row M subtiles.
  // Row r -> image n, patch (pi, pj); lane half selects pw offset 0 or 8.
  int abase[2];
#pragma unroll
  for (int i = 0; i < 2; ++i) {
    int r  = m_base + i * 16 + l15;
    int n  = r / PATCHES;
    int p  = r - n * PATCHES;
    int pi = p / 14;
    int pj = p - pi * 14;
    abase[i] = n * IMG_N + pi * (16 * 224) + pj * 16 + lhalf * 8;
  }

  // B operand: lane l15 owns column n_base+l15 -> row (n_base+l15) of W.
  const _Float16* wrow = Wh + (size_t)(n_base + l15) * HD + lhalf * 8;

  v8f acc[2][4];
#pragma unroll
  for (int i = 0; i < 2; ++i)
#pragma unroll
    for (int j = 0; j < 4; ++j)
      acc[i][j] = {};

  for (int k0 = 0; k0 < HD; k0 += 32) {
    // K index k = c*256 + ph*16 + pw ; within a 32-wide K slab c is constant
    // and the "+16" K chunk is exactly one pixel row (+224 floats) below.
    const int c    = k0 >> 8;
    const int ph   = (k0 & 255) >> 4;
    const int aofs = c * IMG_CH + ph * 224;

    v16h afrag[2];
#pragma unroll
    for (int i = 0; i < 2; ++i) {
      const float* p = img + abase[i] + aofs;          // 16B aligned
      v4f x0 = *reinterpret_cast<const v4f*>(p);       // K +0..3
      v4f x1 = *reinterpret_cast<const v4f*>(p + 4);   // K +4..7
      v4f y0 = *reinterpret_cast<const v4f*>(p + 224); // K +16..19
      v4f y1 = *reinterpret_cast<const v4f*>(p + 228); // K +20..23
      v16h a;
#pragma unroll
      for (int e = 0; e < 4; ++e) {
        a[e]      = (_Float16)x0[e];
        a[e + 4]  = (_Float16)x1[e];
        a[e + 8]  = (_Float16)y0[e];
        a[e + 12] = (_Float16)y1[e];
      }
      afrag[i] = a;
    }

    v16h bfrag[4];
#pragma unroll
    for (int j = 0; j < 4; ++j) {
      const _Float16* q = wrow + (size_t)j * (16 * HD) + k0; // 16B aligned
      v8h lo = *reinterpret_cast<const v8h*>(q);        // K chunk +0..7
      v8h hi = *reinterpret_cast<const v8h*>(q + 16);   // K chunk +16..23
      bfrag[j] = __builtin_shufflevector(lo, hi,
                   0, 1, 2, 3, 4, 5, 6, 7, 8, 9, 10, 11, 12, 13, 14, 15);
    }

#pragma unroll
    for (int i = 0; i < 2; ++i)
#pragma unroll
      for (int j = 0; j < 4; ++j)
        acc[i][j] = __builtin_amdgcn_wmma_f32_16x16x32_f16(
            /*neg_a=*/false, afrag[i], /*neg_b=*/false, bfrag[j],
            /*c_mod=*/(short)0, acc[i][j], /*reuse_a=*/false, /*reuse_b=*/false);
  }

  // Epilogue: C/D layout -> VGPR v holds row (v + 8*lhalf), col = l15.
#pragma unroll
  for (int j = 0; j < 4; ++j) {
    const float bj = bias[n_base + j * 16 + l15];
#pragma unroll
    for (int i = 0; i < 2; ++i) {
      const int row0 = m_base + i * 16 + lhalf * 8;
      float* o = out + (size_t)row0 * HD + (n_base + j * 16 + l15);
#pragma unroll
      for (int v = 0; v < 8; ++v)
        o[(size_t)v * HD] = acc[i][j][v] + bj;
    }
  }
}

extern "C" void kernel_launch(void* const* d_in, const int* in_sizes, int n_in,
                              void* d_out, int out_size, void* d_ws, size_t ws_size,
                              hipStream_t stream) {
  const float* img  = (const float*)d_in[0];   // [256,3,224,224] fp32
  const float* W    = (const float*)d_in[1];   // [768,768] fp32
  const float* bias = (const float*)d_in[2];   // [768] fp32
  float* out        = (float*)d_out;           // [256,196,768] fp32
  _Float16* Wh      = (_Float16*)d_ws;         // 768*768*2 = 1.18 MB scratch

  // 1) Convert W to f16 once (L2-resident for the whole GEMM).
  w_to_f16_kernel<<<(HD * HD) / (256 * 4), 256, 0, stream>>>(W, Wh);

  // 2) Implicit GEMM: grid x = N blocks (6), y = M blocks (392).
  //    x-major launch order keeps the 6 N-blocks of an M-stripe adjacent,
  //    so the A (image) stripe is read from L2 for 5 of its 6 uses.
  dim3 grid(HD / BN, (256 * PATCHES) / BM);
  vit_patch_embed<<<grid, 256, 0, stream>>>(img, Wh, bias, out);
}
